// PerturbedTopK_79980880986196
// MI455X (gfx1250) — compile-verified, matrix operand order checked by reading, not run on hardware
//
#include <hip/hip_runtime.h>

// ---------------------------------------------------------------------------
// PerturbedTopK on gfx1250 (MI455X, wave32).
//   x:     (64, 2048) f32
//   noise: (64, 256, 2048) f32
//   out:   topk (64,2048) f32  ||  1-topk (64,2048) f32
//
// One workgroup = one (b, group of 64 samples). Radix-select (4x8-bit) finds
// the exact 256th-largest value per row; indicators are staged transposed in
// LDS as bytes and column-summed over 64 samples with V_WMMA_I32_16X16X64_IU8
// (A = ones). Partial counts merged with global atomics, finalized separately.
//
// Memory plan: noise (128 MB) is streamed exactly once as global_load_b128
// (two per thread per row), with a global_prefetch of the next sample row
// issued before the barrier-heavy selection phase so the HBM/L2 fill overlaps
// the LDS histogram work. HBM floor ~5.5 us @ 23.3 TB/s.
// ---------------------------------------------------------------------------

#define K_SEL   256
#define NS      256
#define B_DIM   64
#define D_DIM   2048
#define SIGMA   0.05f

#define THREADS            256
#define SAMPLES_PER_WG     64
#define GROUPS_PER_B       (NS / SAMPLES_PER_WG)          // 4
#define ELEMS_PER_THREAD   (D_DIM / THREADS)              // 8
#define N_TILES            (D_DIM / 16)                   // 128

// Dynamic-LDS carve-up (bytes):
#define SM_HIST   0                       // 256 * u32   = 1024
#define SM_SUF    1024                    // 257 * u32   = 1028 -> pad
#define SM_SCAL   2064                    // 2   * u32
#define SM_IND    2080                    // 2048*64 u8  = 131072  (16B aligned)
#define SM_TOTAL  (SM_IND + D_DIM * SAMPLES_PER_WG)

typedef __attribute__((ext_vector_type(8))) int v8i;

__device__ __forceinline__ unsigned order_key(float f) {
  // Order-preserving map: larger float <=> larger unsigned.
  unsigned u = __float_as_uint(f);
  return u ^ ((u >> 31) ? 0xFFFFFFFFu : 0x80000000u);
}

__global__ __launch_bounds__(THREADS) void topk_count_kernel(
    const float* __restrict__ x, const float* __restrict__ noise,
    unsigned* __restrict__ counts) {
  extern __shared__ __align__(16) unsigned char smem[];
  unsigned* hist          = (unsigned*)(smem + SM_HIST);   // [256]
  unsigned* suf           = (unsigned*)(smem + SM_SUF);    // [257]
  unsigned* scal          = (unsigned*)(smem + SM_SCAL);   // [0]=prefix [1]=remaining
  unsigned char* ind_t    = smem + SM_IND;                 // [d][s] bytes

  const unsigned tid = threadIdx.x;
  const unsigned b   = blockIdx.x / GROUPS_PER_B;
  const unsigned g   = blockIdx.x % GROUPS_PER_B;

  // This thread owns 8 contiguous d positions: d0 .. d0+7 (b128-friendly).
  const unsigned d0 = tid * ELEMS_PER_THREAD;

  // x-row slice lives in registers, reused for all 64 samples (2x b128 loads).
  float xr[ELEMS_PER_THREAD];
  {
    const float4* xp = (const float4*)(x + (size_t)b * D_DIM + d0);
    float4 x_lo = xp[0];
    float4 x_hi = xp[1];
    xr[0] = x_lo.x; xr[1] = x_lo.y; xr[2] = x_lo.z; xr[3] = x_lo.w;
    xr[4] = x_hi.x; xr[5] = x_hi.y; xr[6] = x_hi.z; xr[7] = x_hi.w;
  }

  const float* noise_base = noise + ((size_t)(b * NS + g * SAMPLES_PER_WG)) * (size_t)D_DIM;

  for (unsigned sl = 0; sl < SAMPLES_PER_WG; ++sl) {
    const float* nrow = noise_base + (size_t)sl * D_DIM;

    // Two wide loads for this sample's slice.
    const float4* np = (const float4*)(nrow + d0);
    float4 n_lo = np[0];
    float4 n_hi = np[1];

    // Prefetch next sample's slice; the fill overlaps the selection phase.
    if (sl + 1 < SAMPLES_PER_WG)
      __builtin_prefetch(nrow + D_DIM + d0, 0, 1);

    float pv[ELEMS_PER_THREAD];
    pv[0] = n_lo.x; pv[1] = n_lo.y; pv[2] = n_lo.z; pv[3] = n_lo.w;
    pv[4] = n_hi.x; pv[5] = n_hi.y; pv[6] = n_hi.z; pv[7] = n_hi.w;

    unsigned key[ELEMS_PER_THREAD];
#pragma unroll
    for (int j = 0; j < ELEMS_PER_THREAD; ++j)
      key[j] = order_key(fmaf(pv[j], SIGMA, xr[j]));

    if (tid == 0) { scal[0] = 0u; scal[1] = K_SEL; }

    // 4 radix passes, 8 bits each, MSB -> LSB, to find the K-th largest key.
#pragma unroll
    for (int pass = 0; pass < 4; ++pass) {
      const int shift = 24 - 8 * pass;
      hist[tid] = 0u;
      __syncthreads();
      const unsigned prefix = scal[0];
#pragma unroll
      for (int j = 0; j < ELEMS_PER_THREAD; ++j) {
        bool match = (pass == 0) || ((key[j] >> (shift + 8)) == prefix);
        if (match) atomicAdd(&hist[(key[j] >> shift) & 0xFFu], 1u);
      }
      __syncthreads();

      // Parallel suffix sum over 256 bins: suf[i] = sum_{j>=i} hist[j].
      suf[tid] = hist[tid];
      if (tid == 0) suf[256] = 0u;
      __syncthreads();
#pragma unroll
      for (int st = 1; st < 256; st <<= 1) {
        unsigned v   = suf[tid];
        unsigned add = (tid + st <= 255) ? suf[tid + st] : 0u;
        __syncthreads();
        suf[tid] = v + add;
        __syncthreads();
      }
      // Highest bin whose suffix count still covers `remaining`.
      const unsigned remaining = scal[1];
      if (suf[tid] >= remaining && suf[tid + 1] < remaining) {
        scal[0] = (prefix << 8) | tid;      // extend prefix
        scal[1] = remaining - suf[tid + 1]; // rank within selected bin
      }
      __syncthreads();
    }
    const unsigned kth = scal[0];  // exact K-th largest key value

    // Indicator bytes, transposed: ind_t[d * 64 + sl].
#pragma unroll
    for (int j = 0; j < ELEMS_PER_THREAD; ++j) {
      ind_t[(d0 + j) * SAMPLES_PER_WG + sl] =
          (key[j] >= kth) ? (unsigned char)1 : (unsigned char)0;
    }
    __syncthreads();
  }

  // -------------------------------------------------------------------------
  // Sample-sum via WMMA: D = ones(16x64) x ind(64x16) -> every row of D holds
  // the per-column sums over the 64 samples. 128 tiles of 16 d-positions.
  // -------------------------------------------------------------------------
  const unsigned wave = tid >> 5;
  const unsigned lane = tid & 31;
  const unsigned n    = lane & 15;   // B-matrix column this lane holds
  const unsigned half = lane >> 4;   // K half selector (ISA 8-bit B layout)

  v8i a;
#pragma unroll
  for (int i = 0; i < 8; ++i) a[i] = 0x01010101;  // all-ones A matrix (bytes)

  for (unsigned tile = wave; tile < N_TILES; tile += THREADS / 32) {
    const unsigned d = tile * 16 + n;
    const uint4* p = (const uint4*)&ind_t[d * SAMPLES_PER_WG];
    uint4 lo = p[half];       // samples K =  0..15 (lanes 0-15) / 16..31 (16-31)
    uint4 hi = p[2 + half];   // samples K = 32..47 (lanes 0-15) / 48..63 (16-31)
    v8i bm;
    bm[0] = (int)lo.x; bm[1] = (int)lo.y; bm[2] = (int)lo.z; bm[3] = (int)lo.w;
    bm[4] = (int)hi.x; bm[5] = (int)hi.y; bm[6] = (int)hi.z; bm[7] = (int)hi.w;

    v8i c = {};
    c = __builtin_amdgcn_wmma_i32_16x16x64_iu8(
        /*sgn_a=*/false, a, /*sgn_b=*/false, bm, c,
        /*reuse_a=*/false, /*reuse_b=*/false);

    // VGPR0: lanes 0-15 -> (M=0, N=lane): column sums for this tile's 16 d's.
    if (lane < 16)
      atomicAdd(&counts[b * D_DIM + tile * 16 + lane], (unsigned)c[0]);
  }
}

__global__ __launch_bounds__(256) void finalize_kernel(
    const unsigned* __restrict__ counts, float* __restrict__ out) {
  unsigned idx = blockIdx.x * blockDim.x + threadIdx.x;
  if (idx < B_DIM * D_DIM) {
    float t = (float)counts[idx] * (1.0f / (float)NS);
    out[idx] = t;                          // topk indicator mean
    out[B_DIM * D_DIM + idx] = 1.0f - t;   // non-topk mean
  }
}

extern "C" void kernel_launch(void* const* d_in, const int* in_sizes, int n_in,
                              void* d_out, int out_size, void* d_ws, size_t ws_size,
                              hipStream_t stream) {
  const float* x     = (const float*)d_in[0];
  const float* noise = (const float*)d_in[1];
  unsigned* counts   = (unsigned*)d_ws;
  float* out         = (float*)d_out;

  hipMemsetAsync(counts, 0, (size_t)B_DIM * D_DIM * sizeof(unsigned), stream);

  dim3 grid1(B_DIM * GROUPS_PER_B);  // 256 workgroups
  topk_count_kernel<<<grid1, THREADS, SM_TOTAL, stream>>>(x, noise, counts);

  dim3 grid2((B_DIM * D_DIM + 255) / 256);
  finalize_kernel<<<grid2, 256, 0, stream>>>(counts, out);
}
